// GroupedQueryAttention_34144990003779
// MI455X (gfx1250) — matrix operand order repaired and checked by reading, not compiled, not run on hardware
//
#include <hip/hip_runtime.h>
#include <hip/hip_bf16.h>
#include <stdint.h>

#define S_LEN     4096
#define E_DIM     2048
#define H_HEADS   16
#define KVH_HEADS 4
#define D_HEAD    128
#define KV_DIM    512   // KVH*D
#define GQ        4     // H/KVH

typedef __attribute__((ext_vector_type(16))) __bf16 v16bf;
typedef __attribute__((ext_vector_type(8)))  float  v8f;

#if defined(__has_builtin)
#  if __has_builtin(__builtin_amdgcn_tensor_load_to_lds) && \
      __has_builtin(__builtin_amdgcn_s_wait_tensorcnt)
#    define USE_TDM 1
#  endif
#endif
#ifndef USE_TDM
#  define USE_TDM 0
#endif

union Frag {
    v16bf b;
    uint4 q[2];
    unsigned short u[16];
};

__device__ inline unsigned short f2bf(float f) {
    union { float f; unsigned int u; } v; v.f = f;
    unsigned int r = v.u + 0x7FFFu + ((v.u >> 16) & 1u);  // RNE
    return (unsigned short)(r >> 16);
}

__device__ inline v8f v8f_zero() {
    v8f z;
#pragma unroll
    for (int i = 0; i < 8; ++i) z[i] = 0.0f;
    return z;
}

__device__ inline v8f wmma_bf16(const Frag& a, const Frag& b, v8f c) {
    // D = A(16x32 bf16) * B(32x16 bf16) + C(16x16 f32)
    return __builtin_amdgcn_wmma_f32_16x16x32_bf16(
        false, a.b, false, b.b, (short)0, c, false, false);
}

#if USE_TDM
typedef __attribute__((ext_vector_type(4))) unsigned int u32x4;
typedef __attribute__((ext_vector_type(8))) int          i32x8;
typedef __attribute__((ext_vector_type(4))) int          i32x4;

// Flat LDS addresses carry the byte offset in addr[31:0] (aperture in high bits).
__device__ inline unsigned int lds_off(const void* p) {
    return (unsigned int)(unsigned long long)(uintptr_t)p;
}

// TDM 2D tile load, bf16 elements (data_size code 1 = 2 bytes).
// D# built per CDNA5 ISA 8.3/8.4: group0 {count=1, lds_addr, global_addr, type=2},
// group1 {mask=0, data_size, tensor_dim0/1, tile_dim0/1, tensor_dim0_stride}.
__device__ inline void tdm_load_2d_bf16(unsigned int lds_byte_addr,
                                        const unsigned short* gptr,
                                        unsigned int tensor_d0, unsigned int tensor_d1,
                                        unsigned long long stride0,
                                        unsigned int tile_d0, unsigned int tile_d1)
{
    const unsigned long long ga = (unsigned long long)(uintptr_t)gptr;
    u32x4 g0;
    g0[0] = 1u;                                               // count=1, user mode
    g0[1] = lds_byte_addr;                                    // lds_addr [63:32]
    g0[2] = (unsigned int)(ga & 0xFFFFFFFFu);                 // global_addr lo
    g0[3] = (unsigned int)((ga >> 32) & 0x1FFFFFFu) | (2u << 30);  // addr hi | type=2
    i32x8 g1;
    g1[0] = (int)(1u << 16);                                  // wg_mask=0, data_size=1 (2B)
    g1[1] = (int)((tensor_d0 & 0xFFFFu) << 16);               // atomic_addr=0 | dim0[15:0]
    g1[2] = (int)(((tensor_d0 >> 16) & 0xFFFFu) | ((tensor_d1 & 0xFFFFu) << 16));
    g1[3] = (int)(((tensor_d1 >> 16) & 0xFFFFu) | ((tile_d0 & 0xFFFFu) << 16));
    g1[4] = (int)(tile_d1 & 0xFFFFu);                         // tile_dim1 | tile_dim2=0
    g1[5] = (int)(stride0 & 0xFFFFFFFFu);                     // dim0_stride lo
    g1[6] = (int)((stride0 >> 32) & 0xFFFFu);                 // dim0_stride hi | dim1_stride=0
    g1[7] = 0;
    const i32x4 z4 = {0, 0, 0, 0};
#if defined(__clang_major__) && __clang_major__ >= 23
    const i32x8 z8 = {0, 0, 0, 0, 0, 0, 0, 0};
    __builtin_amdgcn_tensor_load_to_lds(g0, g1, z4, z4, z8, 0);
#else
    __builtin_amdgcn_tensor_load_to_lds(g0, g1, z4, z4, 0);
#endif
}
#endif  // USE_TDM

// ---------------------------------------------------------------------------
// C[M,N] = A[M,K] @ W[N,K]^T + bias[N]    (bf16 WMMA, f32 accumulate)
// BK=64: 16 WMMAs per wave per barrier pair.
// MODE 0: bf16 C -> outBf [M,N]                              (Q projection)
// MODE 1: cols <512: bf16 -> outBf (K row-major [S,512]),
//                    f32  -> outF  (K as [KVH,S,D])
//         cols>=512: bf16 -> outBf2 (V transposed [512,S]),
//                    f32  -> outF2 (V as [KVH,S,D])          (KV projection)
// MODE 2: f32 C -> outF [M,N]                                (O projection)
// ---------------------------------------------------------------------------
template<int MODE, bool ABF16>
__global__ __launch_bounds__(256)
void gemm_bf16_wmma(const void* __restrict__ Aptr, const float* __restrict__ W,
                    const float* __restrict__ bias,
                    int M, int N, int K,
                    unsigned short* __restrict__ outBf,
                    unsigned short* __restrict__ outBf2,
                    float* __restrict__ outF,
                    float* __restrict__ outF2)
{
    __shared__ __align__(16) unsigned short As[128 * 80];  // 128 x 64, stride 80
    __shared__ __align__(16) unsigned short Bs[128 * 80];

    const int tid    = threadIdx.x;
    const int lane   = tid & 31;
    const int wid    = tid >> 5;
    const int wave_m = wid >> 2;          // 0..1  (64 rows each)
    const int wave_n = wid & 3;           // 0..3  (32 cols each)
    const int row_blk = blockIdx.y * 128;
    const int col_blk = blockIdx.x * 128;

    const int lrow = tid >> 1;            // 0..127
    const int lcol = (tid & 1) * 32;      // 0 / 32

    const float*          Af = (const float*)Aptr;
    const unsigned short* Ab = (const unsigned short*)Aptr;

    v8f acc[4][2];
#pragma unroll
    for (int i = 0; i < 4; ++i)
#pragma unroll
        for (int j = 0; j < 2; ++j) acc[i][j] = v8f_zero();

    for (int k0 = 0; k0 < K; k0 += 64) {
        // ---- stage A tile (128x64) into LDS as bf16 ----
        if (ABF16) {
            const unsigned short* src = Ab + (size_t)(row_blk + lrow) * K + k0 + lcol;
#pragma unroll
            for (int j = 0; j < 4; ++j)
                *(uint4*)&As[lrow * 80 + lcol + j * 8] = *(const uint4*)(src + j * 8);
        } else {
            const float* src = Af + (size_t)(row_blk + lrow) * K + k0 + lcol;
            if (k0 + 64 < K) __builtin_prefetch(src + 64, 0, 1);
#pragma unroll
            for (int j = 0; j < 8; ++j) {
                float4 f = *(const float4*)(src + j * 4);
                As[lrow * 80 + lcol + j * 4 + 0] = f2bf(f.x);
                As[lrow * 80 + lcol + j * 4 + 1] = f2bf(f.y);
                As[lrow * 80 + lcol + j * 4 + 2] = f2bf(f.z);
                As[lrow * 80 + lcol + j * 4 + 3] = f2bf(f.w);
            }
        }
        // ---- stage W tile (128 output cols x 64 K) ----
        {
            const float* src = W + (size_t)(col_blk + lrow) * K + k0 + lcol;
            if (k0 + 64 < K) __builtin_prefetch(src + 64, 0, 1);
#pragma unroll
            for (int j = 0; j < 8; ++j) {
                float4 f = *(const float4*)(src + j * 4);
                Bs[lrow * 80 + lcol + j * 4 + 0] = f2bf(f.x);
                Bs[lrow * 80 + lcol + j * 4 + 1] = f2bf(f.y);
                Bs[lrow * 80 + lcol + j * 4 + 2] = f2bf(f.z);
                Bs[lrow * 80 + lcol + j * 4 + 3] = f2bf(f.w);
            }
        }
        __syncthreads();

        const int kbA = (lane < 16) ? 0 : 8;
        const int kkB = (lane < 16) ? 0 : 8;
#pragma unroll
        for (int ks = 0; ks < 64; ks += 32) {
            Frag fa[4], fb[2];
#pragma unroll
            for (int mt = 0; mt < 4; ++mt) {
                const int r = wave_m * 64 + mt * 16 + (lane & 15);
                fa[mt].q[0] = *(const uint4*)&As[r * 80 + ks + kbA];
                fa[mt].q[1] = *(const uint4*)&As[r * 80 + ks + kbA + 16];
            }
#pragma unroll
            for (int nt = 0; nt < 2; ++nt) {
                const int c = wave_n * 32 + nt * 16 + (lane & 15);
                fb[nt].q[0] = *(const uint4*)&Bs[c * 80 + ks + kkB];
                fb[nt].q[1] = *(const uint4*)&Bs[c * 80 + ks + kkB + 16];
            }
#pragma unroll
            for (int mt = 0; mt < 4; ++mt)
#pragma unroll
                for (int nt = 0; nt < 2; ++nt)
                    acc[mt][nt] = wmma_bf16(fa[mt], fb[nt], acc[mt][nt]);
        }
        __syncthreads();
    }

    // ---- epilogue ----
    const int hi8 = (lane < 16) ? 0 : 8;
#pragma unroll
    for (int mt = 0; mt < 4; ++mt) {
#pragma unroll
        for (int nt = 0; nt < 2; ++nt) {
#pragma unroll
            for (int r = 0; r < 8; ++r) {
                const int row_g = row_blk + wave_m * 64 + mt * 16 + r + hi8;
                const int col_g = col_blk + wave_n * 32 + nt * 16 + (lane & 15);
                const float v = acc[mt][nt][r] + bias[col_g];
                if (MODE == 0) {
                    outBf[(size_t)row_g * N + col_g] = f2bf(v);
                } else if (MODE == 1) {
                    if (col_g < KV_DIM) {
                        outBf[(size_t)row_g * KV_DIM + col_g] = f2bf(v);
                        outF[(size_t)(col_g >> 7) * S_LEN * D_HEAD +
                             (size_t)row_g * D_HEAD + (col_g & 127)] = v;
                    } else {
                        const int nv = col_g - KV_DIM;
                        outBf2[(size_t)nv * S_LEN + row_g] = f2bf(v);
                        outF2[(size_t)(nv >> 7) * S_LEN * D_HEAD +
                              (size_t)row_g * D_HEAD + (nv & 127)] = v;
                    }
                } else {
                    outF[(size_t)row_g * N + col_g] = v;
                }
            }
        }
    }
}

// ---------------------------------------------------------------------------
// Causal flash attention, GQA. Block = 8 waves = 128 query rows of one head.
// K/V 32-key tiles are block-shared, double-buffered in LDS, staged by the
// Tensor Data Mover (wave 0 issues tensor_load_to_lds, waits TENSORcnt, then a
// workgroup barrier releases all waves); prefetch of tile i+1 overlaps the
// 16 WMMAs + online softmax of tile i. Grid = (S/128, H).
// ---------------------------------------------------------------------------
__global__ __launch_bounds__(256)
void gqa_flash_attn(const unsigned short* __restrict__ Qbf,
                    const unsigned short* __restrict__ Kbf,
                    const unsigned short* __restrict__ Vt,
                    unsigned short* __restrict__ Obf)
{
    __shared__ __align__(16) unsigned short Kst[2][32 * 128];  // [key][d]
    __shared__ __align__(16) unsigned short Vst[2][128 * 32];  // [d][key]
    __shared__ __align__(16) unsigned short Plds[8][16 * 48];

    const int tid  = threadIdx.x;
    const int lane = tid & 31;
    const int wid  = tid >> 5;
    const int h    = blockIdx.y;
    const int kvh  = h / GQ;
    const int q0   = blockIdx.x * 128 + wid * 16;
    const int l16  = lane & 15;
    const int hi8  = (lane < 16) ? 0 : 8;
    const int kbA  = (lane < 16) ? 0 : 8;
    const int kk16 = (lane < 16) ? 0 : 16;
    const int n_tiles = blockIdx.x * 4 + 4;   // keys 0 .. blk*128+127

    // Q fragments: 16 rows x 128, kept in registers for the whole key loop
    Frag fq[4];
    {
        const int qrow = q0 + l16;
#pragma unroll
        for (int ds = 0; ds < 4; ++ds) {
            const unsigned short* p =
                Qbf + (size_t)qrow * E_DIM + h * D_HEAD + ds * 32 + kbA;
            fq[ds].q[0] = *(const uint4*)p;
            fq[ds].q[1] = *(const uint4*)(p + 16);
        }
    }

    float mrow[8], lrowv[8];
    v8f o[8];
#pragma unroll
    for (int r = 0; r < 8; ++r) { mrow[r] = -3.0e38f; lrowv[r] = 0.0f; }
#pragma unroll
    for (int t = 0; t < 8; ++t) o[t] = v8f_zero();

    auto stage = [&](int tile, int buf) {
        const int kb0 = tile * 32;
#if USE_TDM
        if (wid == 0) {
            // K tile: 32 keys x 128 d   (row stride KV_DIM in the K buffer)
            tdm_load_2d_bf16(lds_off(&Kst[buf][0]),
                             Kbf + (size_t)kb0 * KV_DIM + kvh * D_HEAD,
                             KV_DIM, S_LEN, KV_DIM, D_HEAD, 32);
            // V tile: 128 d x 32 keys   (row stride S in the V^T buffer)
            tdm_load_2d_bf16(lds_off(&Vst[buf][0]),
                             Vt + (size_t)kvh * D_HEAD * S_LEN + kb0,
                             S_LEN, KV_DIM, S_LEN, 32, D_HEAD);
        }
#else
        {   // cooperative fallback copy, 256 threads
            const int kr = tid >> 3, kc = (tid & 7) * 16;
            const unsigned short* ks =
                Kbf + (size_t)(kb0 + kr) * KV_DIM + kvh * D_HEAD + kc;
            *(uint4*)&Kst[buf][kr * 128 + kc]     = *(const uint4*)ks;
            *(uint4*)&Kst[buf][kr * 128 + kc + 8] = *(const uint4*)(ks + 8);
            const int vr = tid >> 1, vc = (tid & 1) * 16;
            const unsigned short* vs =
                Vt + (size_t)(kvh * D_HEAD + vr) * S_LEN + kb0 + vc;
            *(uint4*)&Vst[buf][vr * 32 + vc]     = *(const uint4*)vs;
            *(uint4*)&Vst[buf][vr * 32 + vc + 8] = *(const uint4*)(vs + 8);
        }
#endif
    };

    const float sc = 0.08838834764831845f;  // 1/sqrt(128)
    const int kmax = q0 + 15;

    stage(0, 0);
    for (int it = 0; it < n_tiles; ++it) {
#if USE_TDM
        if (wid == 0) __builtin_amdgcn_s_wait_tensorcnt((unsigned short)0);
#endif
        __syncthreads();                       // tile `it` visible to all waves
        if (it + 1 < n_tiles) stage(it + 1, (it + 1) & 1);  // overlap DMA

        const int kb0 = it * 32;
        const int buf = it & 1;
        if (kb0 <= kmax) {                      // wave-uniform causal skip
            // ---- scores: Q(16x128) x K^T(128x32) -> two 16x16 tiles ----
            v8f s0 = v8f_zero(), s1 = v8f_zero();
#pragma unroll
            for (int ds = 0; ds < 4; ++ds) {
                Frag b0, b1;
                const unsigned short* pk0 = &Kst[buf][l16 * 128 + ds * 32 + kbA];
                const unsigned short* pk1 = &Kst[buf][(16 + l16) * 128 + ds * 32 + kbA];
                b0.q[0] = *(const uint4*)pk0;  b0.q[1] = *(const uint4*)(pk0 + 16);
                b1.q[0] = *(const uint4*)pk1;  b1.q[1] = *(const uint4*)(pk1 + 16);
                s0 = wmma_bf16(fq[ds], b0, s0);
                s1 = wmma_bf16(fq[ds], b1, s1);
            }

            // ---- online softmax: rows live in one VGPR across a 16-lane
            //      half -> shfl_xor {1,2,4,8} row reductions ----
            float alpha[8];
#pragma unroll
            for (int r = 0; r < 8; ++r) {
                float x0 = s0[r] * sc;
                float x1 = s1[r] * sc;
                const int qr  = q0 + r + hi8;
                const int kg0 = kb0 + l16;
                if (kg0 > qr)      x0 = -3.0e38f;   // causal (lane select)
                if (kg0 + 16 > qr) x1 = -3.0e38f;
                float mx = fmaxf(x0, x1);
                mx = fmaxf(mx, __shfl_xor(mx, 1));
                mx = fmaxf(mx, __shfl_xor(mx, 2));
                mx = fmaxf(mx, __shfl_xor(mx, 4));
                mx = fmaxf(mx, __shfl_xor(mx, 8));
                const float mn = fmaxf(mrow[r], mx);
                const float a  = __expf(mrow[r] - mn);
                const float p0 = __expf(x0 - mn);
                const float p1 = __expf(x1 - mn);
                float rs = p0 + p1;
                rs += __shfl_xor(rs, 1);
                rs += __shfl_xor(rs, 2);
                rs += __shfl_xor(rs, 4);
                rs += __shfl_xor(rs, 8);
                lrowv[r] = lrowv[r] * a + rs;
                mrow[r]  = mn;
                alpha[r] = a;
                const int prow = r + hi8;       // C layout -> LDS row-major P
                Plds[wid][prow * 48 + l16]      = f2bf(p0);
                Plds[wid][prow * 48 + 16 + l16] = f2bf(p1);
            }

            // ---- rescale running O ----
#pragma unroll
            for (int t = 0; t < 8; ++t)
#pragma unroll
                for (int r = 0; r < 8; ++r) o[t][r] *= alpha[r];

            // ---- reload P in A-fragment layout (wave-private LDS) ----
            Frag fp;
            fp.q[0] = *(const uint4*)&Plds[wid][l16 * 48 + kbA];
            fp.q[1] = *(const uint4*)&Plds[wid][l16 * 48 + kbA + 16];

            // ---- O += P(16x32) x V(32x128) ----
#pragma unroll
            for (int t = 0; t < 8; ++t) {
                Frag bv;
                const unsigned short* pv = &Vst[buf][(t * 16 + l16) * 32 + kbA];
                bv.q[0] = *(const uint4*)pv;  bv.q[1] = *(const uint4*)(pv + 16);
                o[t] = wmma_bf16(fp, bv, o[t]);
            }
        }
        // buffer reuse protected by the barrier at the top of the next iter
    }

    // ---- finalize: O /= l, store bf16 [S, h*D + d] for the O projection ----
    float inv[8];
#pragma unroll
    for (int r = 0; r < 8; ++r) inv[r] = 1.0f / lrowv[r];
#pragma unroll
    for (int t = 0; t < 8; ++t) {
#pragma unroll
        for (int r = 0; r < 8; ++r) {
            const float v  = o[t][r] * inv[r];
            const int row  = q0 + r + hi8;
            const int col  = h * D_HEAD + t * 16 + l16;
            Obf[(size_t)row * E_DIM + col] = f2bf(v);
        }
    }
}

// ---------------------------------------------------------------------------
extern "C" void kernel_launch(void* const* d_in, const int* in_sizes, int n_in,
                              void* d_out, int out_size, void* d_ws, size_t ws_size,
                              hipStream_t stream)
{
    const float* seq   = (const float*)d_in[0];
    // d_in[1] = mask (unused: causality computed analytically)
    const float* Wq_w  = (const float*)d_in[2];
    const float* Wq_b  = (const float*)d_in[3];
    const float* Wkv_w = (const float*)d_in[4];
    const float* Wkv_b = (const float*)d_in[5];
    const float* Wo_w  = (const float*)d_in[6];
    const float* Wo_b  = (const float*)d_in[7];

    float* outO = (float*)d_out;                                  // [S,E]
    float* outK = outO + (size_t)S_LEN * E_DIM;                   // [KVH,S,D]
    float* outV = outK + (size_t)KVH_HEADS * S_LEN * D_HEAD;      // [KVH,S,D]

    unsigned short* Qbf = (unsigned short*)d_ws;                  // [S,E] bf16
    unsigned short* Kbf = Qbf + (size_t)S_LEN * E_DIM;            // [S,512] bf16
    unsigned short* Vt  = Kbf + (size_t)S_LEN * KV_DIM;           // [512,S] bf16
    unsigned short* Obf = Vt  + (size_t)KV_DIM * S_LEN;           // [S,E] bf16

    const dim3 blk(256);

    // Q = seq @ Wq^T + b            (f32 in, bf16 out)
    gemm_bf16_wmma<0, false><<<dim3(E_DIM / 128, S_LEN / 128), blk, 0, stream>>>(
        seq, Wq_w, Wq_b, S_LEN, E_DIM, E_DIM, Qbf, nullptr, nullptr, nullptr);

    // KV = seq @ Wkv^T + b          (K bf16 row-major, V bf16 transposed,
    //                                K/V f32 straight into d_out)
    gemm_bf16_wmma<1, false><<<dim3((2 * KV_DIM) / 128, S_LEN / 128), blk, 0, stream>>>(
        seq, Wkv_w, Wkv_b, S_LEN, 2 * KV_DIM, E_DIM, Kbf, Vt, outK, outV);

    // flash attention (causal, GQA, TDM-staged K/V tiles)
    gqa_flash_attn<<<dim3(S_LEN / 128, H_HEADS), blk, 0, stream>>>(Qbf, Kbf, Vt, Obf);

    // out = O @ Wo^T + b            (bf16 in, f32 out)
    gemm_bf16_wmma<2, true><<<dim3(E_DIM / 128, S_LEN / 128), blk, 0, stream>>>(
        Obf, Wo_w, Wo_b, S_LEN, E_DIM, E_DIM, nullptr, nullptr, outO, nullptr);
}
